// FeedForwardTransformer_83279415869580
// MI455X (gfx1250) — compile-verified, matrix operand order checked by reading, not run on hardware
//
#include <hip/hip_runtime.h>
#include <hip/hip_bf16.h>
#include <cstdint>

#define B_   8
#define S_   2048
#define D_   256
#define H_   2
#define DK_  128
#define DH_  1024
#define MTOT (B_ * S_)

#define USE_ASYNC_COPY 1

typedef __attribute__((ext_vector_type(16))) __bf16 v16bf;
typedef __attribute__((ext_vector_type(8)))  float  v8f;

__device__ __forceinline__ v8f wmma_bf16(v16bf a, v16bf b, v8f c) {
  // D = A(16x32) * B(32x16) + C, f32 accumulate
  return __builtin_amdgcn_wmma_f32_16x16x32_bf16(false, a, false, b, (short)0, c,
                                                 false, false);
}

__device__ __forceinline__ v8f vzero8() {
  v8f z;
#pragma unroll
  for (int i = 0; i < 8; ++i) z[i] = 0.0f;
  return z;
}

// 16(entity) x 32(K) fragment from a row-major tile (K contiguous, leading dim ld).
__device__ __forceinline__ v16bf load_frag_rm(const __bf16* base, int ld) {
  const int lane = threadIdx.x & 31;
  const __bf16* p = base + (lane & 15) * ld + ((lane >> 4) << 3);
  v16bf f;
#pragma unroll
  for (int e = 0; e < 8; ++e) f[e] = p[e];
#pragma unroll
  for (int e = 0; e < 8; ++e) f[e + 8] = p[e + 16];
  return f;
}

// Fragment from a K-major tile: element [k][n], leading dim ld.
__device__ __forceinline__ v16bf load_frag_km(const __bf16* base, int ld) {
  const int lane = threadIdx.x & 31;
  const __bf16* p = base + (((lane >> 4) << 3)) * ld + (lane & 15);
  v16bf f;
#pragma unroll
  for (int e = 0; e < 8; ++e) f[e] = p[e * ld];
#pragma unroll
  for (int e = 0; e < 8; ++e) f[e + 8] = p[(e + 16) * ld];
  return f;
}

#if USE_ASYNC_COPY
// 32-bit LDS offset of a generic pointer into shared memory
__device__ __forceinline__ unsigned lds_off(const void* p) {
  return (unsigned)(unsigned long long)(__attribute__((address_space(3))) const void*)p;
}
// GLOBAL_LOAD_ASYNC_TO_LDS_B128: memory -> LDS, tracked by ASYNCcnt (no VGPR data)
__device__ __forceinline__ void async_copy_b128(unsigned lds_dst, const void* gsrc) {
  asm volatile("global_load_async_to_lds_b128 %0, %1, off"
               :: "v"(lds_dst), "v"(gsrc)
               : "memory");
}
__device__ __forceinline__ void wait_asynccnt0() {
  asm volatile("s_wait_asynccnt 0" ::: "memory");
}
#endif

// ---------------------------------------------------------------- prep kernels
__global__ __launch_bounds__(256) void convert_bf16_kernel(
    const float* __restrict__ src, __bf16* __restrict__ dst, int n) {
  int i = blockIdx.x * blockDim.x + threadIdx.x;
  const int stride = gridDim.x * blockDim.x;
  for (; i < n; i += stride) dst[i] = (__bf16)src[i];
}

__global__ __launch_bounds__(256) void conv1_rearrange_kernel(
    const float* __restrict__ src, __bf16* __restrict__ dst) {
  // src: [DH][D][9] -> dst: [9][DH][D]
  const int total = 9 * DH_ * D_;
  int i = blockIdx.x * blockDim.x + threadIdx.x;
  const int stride = gridDim.x * blockDim.x;
  for (; i < total; i += stride) {
    int tap = i / (DH_ * D_);
    int rem = i - tap * (DH_ * D_);          // rem = o*D + c
    dst[i] = (__bf16)src[(size_t)rem * 9 + tap];
  }
}

// -------------------------------------------------- generic multi-tap NT GEMM
// Out[m,n] = post( alpha * (sum_tap sum_k A[m + tap + shift0, k] * W[tap][n,k] + bias[n]) )
// Block tile 64x256, 8 waves of 32x64, double-buffered LDS, async weight staging.
__global__ __launch_bounds__(256) void gemm_taps_kernel(
    const __bf16* __restrict__ A, const __bf16* __restrict__ W,
    const float* __restrict__ bias, __bf16* __restrict__ Out,
    int Nout, int ntaps, int shift0, float alpha, int do_relu) {
  __shared__ __align__(16) __bf16 sA[2][64 * 32];    // 8 KB
  __shared__ __align__(16) __bf16 sB[2][256 * 32];   // 32 KB
  const int tid  = threadIdx.x;
  const int wave = tid >> 5;
  const int lane = tid & 31;
  const int m0 = blockIdx.x * 64;
  const int n0 = blockIdx.y * 256;
  const int wm = (wave >> 2) * 32;
  const int wn = (wave & 3) * 64;
  v8f acc[2][4];
#pragma unroll
  for (int i = 0; i < 2; ++i)
#pragma unroll
    for (int j = 0; j < 4; ++j) acc[i][j] = vzero8();

  const int nsteps = ntaps * 8;
  auto stage = [&](int step, int buf) {
    const int tap = step >> 3, kc = step & 7;
    const int shift = shift0 + tap;
    {  // A 64x32: predicated (sequence-edge zero fill), sync path
      int row = tid >> 2, off = (tid & 3) * 8;
      int m = m0 + row;
      int sp = (m & (S_ - 1)) + shift;
      uint4 val = {0u, 0u, 0u, 0u};
      if (sp >= 0 && sp < S_)
        val = *(const uint4*)(A + (size_t)(m + shift) * 256 + kc * 32 + off);
      *(uint4*)(&sA[buf][row * 32 + off]) = val;
    }
    const __bf16* Wt = W + ((size_t)tap * Nout + n0) * 256 + kc * 32;
#pragma unroll
    for (int it = 0; it < 4; ++it) {  // B 256x32
      int idx = tid + it * 256;
      int row = idx >> 2, off = (idx & 3) * 8;
#if USE_ASYNC_COPY
      async_copy_b128(lds_off(&sB[buf][row * 32 + off]),
                      Wt + (size_t)row * 256 + off);
#else
      *(uint4*)(&sB[buf][row * 32 + off]) =
          *(const uint4*)(Wt + (size_t)row * 256 + off);
#endif
    }
  };

  stage(0, 0);
#if USE_ASYNC_COPY
  wait_asynccnt0();
#endif
  __syncthreads();
  for (int step = 0; step < nsteps; ++step) {
    const int buf = step & 1;
    if (step + 1 < nsteps) stage(step + 1, buf ^ 1);  // overlap with WMMA below
    const __bf16* a0 = &sA[buf][0];
    const __bf16* b0 = &sB[buf][0];
#pragma unroll
    for (int i = 0; i < 2; ++i) {
      v16bf a = load_frag_rm(a0 + (wm + i * 16) * 32, 32);
#pragma unroll
      for (int j = 0; j < 4; ++j) {
        v16bf b = load_frag_rm(b0 + (wn + j * 16) * 32, 32);
        acc[i][j] = wmma_bf16(a, b, acc[i][j]);
      }
    }
#if USE_ASYNC_COPY
    wait_asynccnt0();
#endif
    __syncthreads();
  }

  const int cl = lane & 15, rh = (lane >> 4) << 3;
#pragma unroll
  for (int i = 0; i < 2; ++i)
#pragma unroll
    for (int j = 0; j < 4; ++j) {
      int col = n0 + wn + j * 16 + cl;
      float bc = bias[col];
#pragma unroll
      for (int r = 0; r < 8; ++r) {
        int row = m0 + wm + i * 16 + rh + r;
        float v = (acc[i][j][r] + bc) * alpha;
        if (do_relu) v = fmaxf(v, 0.0f);
        Out[(size_t)row * Nout + col] = (__bf16)v;
      }
    }
}

// ---------------------------------------------------- flash attention (per b,h)
// Double-buffered async K/V staging; per-wave 16-query tile; WMMA QK^T and PV.
__global__ __launch_bounds__(128) void attn_kernel(
    const __bf16* __restrict__ Q, const __bf16* __restrict__ Km,
    const __bf16* __restrict__ Vm, const unsigned char* __restrict__ mask,
    __bf16* __restrict__ Ctx) {
  __shared__ __align__(16) __bf16 sK[2][64 * 128];   // 32 KB
  __shared__ __align__(16) __bf16 sV[2][64 * 128];   // 32 KB
  __shared__ __align__(16) __bf16 sP[4][16 * 64];    // 8 KB
  __shared__ float sMB[2][64];
  const int tid = threadIdx.x, wave = tid >> 5, lane = tid & 31;
  const int qt = blockIdx.x, h = blockIdx.y, b = blockIdx.z;
  const int cl = lane & 15, rh = (lane >> 4) << 3;
  const size_t qrow0 = (size_t)b * S_ + qt * 64 + wave * 16;
  const int NKT = S_ / 64;

  v16bf qf[4];
#pragma unroll
  for (int kk = 0; kk < 4; ++kk)
    qf[kk] = load_frag_rm(Q + qrow0 * D_ + h * DK_ + kk * 32, D_);

  v8f o[8];
#pragma unroll
  for (int n8 = 0; n8 < 8; ++n8) o[n8] = vzero8();
  float mrow[8], lrow[8];
#pragma unroll
  for (int r = 0; r < 8; ++r) { mrow[r] = -INFINITY; lrow[r] = 0.0f; }

  auto stage = [&](int kt, int buf) {
#pragma unroll
    for (int it = 0; it < 8; ++it) {  // K,V 64x128 each
      int idx = tid + it * 128;
      int row = idx >> 4;
      int off = (idx & 15) * 8;
      size_t g = ((size_t)b * S_ + kt * 64 + row) * D_ + h * DK_ + off;
#if USE_ASYNC_COPY
      async_copy_b128(lds_off(&sK[buf][row * 128 + off]), Km + g);
      async_copy_b128(lds_off(&sV[buf][row * 128 + off]), Vm + g);
#else
      *(uint4*)(&sK[buf][row * 128 + off]) = *(const uint4*)(Km + g);
      *(uint4*)(&sV[buf][row * 128 + off]) = *(const uint4*)(Vm + g);
#endif
    }
    if (tid < 64)
      sMB[buf][tid] = mask[(size_t)b * S_ + kt * 64 + tid] ? -3.0e38f : 0.0f;
  };

  stage(0, 0);
#if USE_ASYNC_COPY
  wait_asynccnt0();
#endif
  __syncthreads();

  for (int kt = 0; kt < NKT; ++kt) {
    const int buf = kt & 1;
    if (kt + 1 < NKT) stage(kt + 1, buf ^ 1);  // async: overlaps compute below
    // warm L2 two tiles ahead (global_prefetch_b8)
    if (kt + 2 < NKT) {
      int prow = tid & 63;
      const __bf16* src = (tid < 64 ? Km : Vm) +
                          ((size_t)b * S_ + (kt + 2) * 64 + prow) * D_ + h * DK_;
      __builtin_prefetch(src, 0, 0);
      __builtin_prefetch(src + 64, 0, 0);
    }

    v8f sc[4];
#pragma unroll
    for (int nn = 0; nn < 4; ++nn) sc[nn] = vzero8();
#pragma unroll
    for (int nn = 0; nn < 4; ++nn)
#pragma unroll
      for (int kk = 0; kk < 4; ++kk) {
        v16bf bfrag = load_frag_rm(&sK[buf][(nn * 16) * 128 + kk * 32], 128);
        sc[nn] = wmma_bf16(qf[kk], bfrag, sc[nn]);
      }

    __bf16* Pw = &sP[wave][0];
#pragma unroll
    for (int r = 0; r < 8; ++r) {
      float mx = -INFINITY;
#pragma unroll
      for (int nn = 0; nn < 4; ++nn) {
        float x = sc[nn][r] + sMB[buf][nn * 16 + cl];
        sc[nn][r] = x;
        mx = fmaxf(mx, x);
      }
#pragma unroll
      for (int d2 = 1; d2 < 16; d2 <<= 1) mx = fmaxf(mx, __shfl_xor(mx, d2));
      float mnew = fmaxf(mrow[r], mx);
      float corr = __expf(mrow[r] - mnew);
      mrow[r] = mnew;
      float rs = 0.0f;
#pragma unroll
      for (int nn = 0; nn < 4; ++nn) {
        float p = __expf(sc[nn][r] - mnew);
        rs += p;
        Pw[(rh + r) * 64 + nn * 16 + cl] = (__bf16)p;  // C-layout -> A-layout
      }
#pragma unroll
      for (int d2 = 1; d2 < 16; d2 <<= 1) rs += __shfl_xor(rs, d2);
      lrow[r] = lrow[r] * corr + rs;
#pragma unroll
      for (int n8 = 0; n8 < 8; ++n8) o[n8][r] *= corr;
    }
    asm volatile("" ::: "memory");  // keep LDS P stores before the reads below

#pragma unroll
    for (int n8 = 0; n8 < 8; ++n8)
#pragma unroll
      for (int kk2 = 0; kk2 < 2; ++kk2) {
        v16bf a = load_frag_rm(Pw + kk2 * 32, 64);
        v16bf bfr = load_frag_km(&sV[buf][(kk2 * 32) * 128 + n8 * 16], 128);
        o[n8] = wmma_bf16(a, bfr, o[n8]);
      }
#if USE_ASYNC_COPY
    wait_asynccnt0();
#endif
    __syncthreads();
  }

#pragma unroll
  for (int r = 0; r < 8; ++r) {
    float inv = 1.0f / lrow[r];
    size_t row = (size_t)b * S_ + qt * 64 + wave * 16 + rh + r;
#pragma unroll
    for (int n8 = 0; n8 < 8; ++n8) {
      int col = h * DK_ + n8 * 16 + cl;
      Ctx[row * D_ + col] = (__bf16)(o[n8][r] * inv);
    }
  }
}

// ----------------------------- GEMM (N=256 full width) + bias + residual + LN
__global__ __launch_bounds__(256) void gemm_ln_kernel(
    const __bf16* __restrict__ A, const __bf16* __restrict__ W,
    const float* __restrict__ bias, const float* __restrict__ resid,
    const float* __restrict__ gam, const float* __restrict__ bet,
    const unsigned char* __restrict__ mask, float* __restrict__ OutF,
    __bf16* __restrict__ OutB, int Kdim) {
  __shared__ __align__(16) __bf16 sA[2][64 * 32];    // 8 KB
  __shared__ __align__(16) __bf16 sB[2][256 * 32];   // 32 KB
  __shared__ __align__(16) float  sC[64 * 256];      // 64 KB
  const int tid = threadIdx.x, wave = tid >> 5, lane = tid & 31;
  const int cl = lane & 15, rh = (lane >> 4) << 3;
  const int m0 = blockIdx.x * 64;
  const int wm = (wave >> 2) * 32;
  const int wn = (wave & 3) * 64;
  v8f acc[2][4];
#pragma unroll
  for (int i = 0; i < 2; ++i)
#pragma unroll
    for (int j = 0; j < 4; ++j) acc[i][j] = vzero8();

  const int nsteps = Kdim / 32;
  auto stage = [&](int step, int buf) {
    const int k0 = step * 32;
    {
      int row = tid >> 2, off = (tid & 3) * 8;
      const void* src = A + (size_t)(m0 + row) * Kdim + k0 + off;
#if USE_ASYNC_COPY
      async_copy_b128(lds_off(&sA[buf][row * 32 + off]), src);
#else
      *(uint4*)(&sA[buf][row * 32 + off]) = *(const uint4*)src;
#endif
    }
#pragma unroll
    for (int it = 0; it < 4; ++it) {
      int idx = tid + it * 256;
      int row = idx >> 2, off = (idx & 3) * 8;
      const void* src = W + (size_t)row * Kdim + k0 + off;
#if USE_ASYNC_COPY
      async_copy_b128(lds_off(&sB[buf][row * 32 + off]), src);
#else
      *(uint4*)(&sB[buf][row * 32 + off]) = *(const uint4*)src;
#endif
    }
  };

  stage(0, 0);
#if USE_ASYNC_COPY
  wait_asynccnt0();
#endif
  __syncthreads();
  for (int step = 0; step < nsteps; ++step) {
    const int buf = step & 1;
    if (step + 1 < nsteps) stage(step + 1, buf ^ 1);
#pragma unroll
    for (int i = 0; i < 2; ++i) {
      v16bf a = load_frag_rm(&sA[buf][(wm + i * 16) * 32], 32);
#pragma unroll
      for (int j = 0; j < 4; ++j) {
        v16bf bb = load_frag_rm(&sB[buf][(wn + j * 16) * 32], 32);
        acc[i][j] = wmma_bf16(a, bb, acc[i][j]);
      }
    }
#if USE_ASYNC_COPY
    wait_asynccnt0();
#endif
    __syncthreads();
  }

  // epilogue: bias + residual into LDS (full 64x256 rows for LN)
#pragma unroll
  for (int i = 0; i < 2; ++i)
#pragma unroll
    for (int j = 0; j < 4; ++j) {
      int col = wn + j * 16 + cl;
      float bc = bias[col];
#pragma unroll
      for (int r = 0; r < 8; ++r) {
        int rl = wm + i * 16 + rh + r;
        sC[rl * 256 + col] =
            acc[i][j][r] + bc + resid[(size_t)(m0 + rl) * 256 + col];
      }
    }
  __syncthreads();

  // layernorm: 4 threads per row, 64 cols each
  {
    int row = tid >> 2, seg = (tid & 3) * 64;
    const float* rp = sC + row * 256;
    float s1 = 0.0f, s2 = 0.0f;
#pragma unroll
    for (int i2 = 0; i2 < 64; ++i2) {
      float x = rp[seg + i2];
      s1 += x;
      s2 += x * x;
    }
    s1 += __shfl_xor(s1, 1); s1 += __shfl_xor(s1, 2);
    s2 += __shfl_xor(s2, 1); s2 += __shfl_xor(s2, 2);
    float mu = s1 * (1.0f / 256.0f);
    float var = s2 * (1.0f / 256.0f) - mu * mu;
    float rstd = rsqrtf(var + 1e-5f);
    int m = m0 + row;
    float keep = mask[m] ? 0.0f : 1.0f;
    size_t obase = (size_t)m * 256;
#pragma unroll
    for (int i2 = 0; i2 < 64; ++i2) {
      int col = seg + i2;
      float y = ((rp[col] - mu) * rstd * gam[col] + bet[col]) * keep;
      OutF[obase + col] = y;
      if (OutB) OutB[obase + col] = (__bf16)y;
    }
  }
}

// ------------------------------------------------------------------- launcher
extern "C" void kernel_launch(void* const* d_in, const int* in_sizes, int n_in,
                              void* d_out, int out_size, void* d_ws,
                              size_t ws_size, hipStream_t stream) {
  (void)in_sizes; (void)n_in; (void)out_size; (void)ws_size;
  const float* batch = (const float*)d_in[0];
  const unsigned char* mask = (const unsigned char*)d_in[1];
  const float* wq = (const float*)d_in[2];
  const float* bq = (const float*)d_in[3];
  const float* wk = (const float*)d_in[4];
  const float* bk = (const float*)d_in[5];
  const float* wv = (const float*)d_in[6];
  const float* bv = (const float*)d_in[7];
  const float* wo = (const float*)d_in[8];
  const float* bo = (const float*)d_in[9];
  const float* ln1g = (const float*)d_in[10];
  const float* ln1b = (const float*)d_in[11];
  const float* c1w = (const float*)d_in[12];
  const float* c1b = (const float*)d_in[13];
  const float* c2w = (const float*)d_in[14];
  const float* c2b = (const float*)d_in[15];
  const float* ln2g = (const float*)d_in[16];
  const float* ln2b = (const float*)d_in[17];

  char* ws = (char*)d_ws;
  const size_t NB = (size_t)MTOT * D_;            // 4,194,304 elements
  size_t o = 0;
  __bf16* batch_bf = (__bf16*)(ws + o); o += 2 * NB;
  __bf16* wq_bf = (__bf16*)(ws + o); o += 2 * (size_t)D_ * D_;
  __bf16* wk_bf = (__bf16*)(ws + o); o += 2 * (size_t)D_ * D_;
  __bf16* wv_bf = (__bf16*)(ws + o); o += 2 * (size_t)D_ * D_;
  __bf16* wo_bf = (__bf16*)(ws + o); o += 2 * (size_t)D_ * D_;
  __bf16* q_bf   = (__bf16*)(ws + o); o += 2 * NB;
  __bf16* k_bf   = (__bf16*)(ws + o); o += 2 * NB;
  __bf16* v_bf   = (__bf16*)(ws + o); o += 2 * NB;
  __bf16* ctx_bf = (__bf16*)(ws + o); o += 2 * NB;
  float*  seq_f  = (float*)(ws + o);  o += 4 * NB;
  __bf16* seq_bf = (__bf16*)(ws + o); o += 2 * NB;
  __bf16* w1_bf  = (__bf16*)(ws + o); o += 2 * (size_t)9 * DH_ * D_;
  __bf16* w2_bf  = (__bf16*)(ws + o); o += 2 * (size_t)D_ * DH_;
  __bf16* h_bf   = (__bf16*)(ws + o); o += 2 * (size_t)MTOT * DH_;

  // ---- prep: bf16 conversion + conv1 weight rearrange
  convert_bf16_kernel<<<1024, 256, 0, stream>>>(batch, batch_bf, (int)NB);
  convert_bf16_kernel<<<64, 256, 0, stream>>>(wq, wq_bf, D_ * D_);
  convert_bf16_kernel<<<64, 256, 0, stream>>>(wk, wk_bf, D_ * D_);
  convert_bf16_kernel<<<64, 256, 0, stream>>>(wv, wv_bf, D_ * D_);
  convert_bf16_kernel<<<64, 256, 0, stream>>>(wo, wo_bf, D_ * D_);
  convert_bf16_kernel<<<256, 256, 0, stream>>>(c2w, w2_bf, D_ * DH_);
  conv1_rearrange_kernel<<<2048, 256, 0, stream>>>(c1w, w1_bf);

  // ---- QKV projections (alpha folds 1/sqrt(D) into Q)
  gemm_taps_kernel<<<dim3(MTOT / 64, 1), 256, 0, stream>>>(
      batch_bf, wq_bf, bq, q_bf, D_, 1, 0, 0.0625f, 0);
  gemm_taps_kernel<<<dim3(MTOT / 64, 1), 256, 0, stream>>>(
      batch_bf, wk_bf, bk, k_bf, D_, 1, 0, 1.0f, 0);
  gemm_taps_kernel<<<dim3(MTOT / 64, 1), 256, 0, stream>>>(
      batch_bf, wv_bf, bv, v_bf, D_, 1, 0, 1.0f, 0);

  // ---- flash attention
  attn_kernel<<<dim3(S_ / 64, H_, B_), 128, 0, stream>>>(q_bf, k_bf, v_bf, mask,
                                                         ctx_bf);

  // ---- O-proj + residual + LN1 + mask  (f32 + bf16 copies of seq)
  gemm_ln_kernel<<<dim3(MTOT / 64), 256, 0, stream>>>(
      ctx_bf, wo_bf, bo, batch, ln1g, ln1b, mask, seq_f, seq_bf, D_);

  // ---- conv1 as 9 accumulated GEMMs + ReLU
  gemm_taps_kernel<<<dim3(MTOT / 64, DH_ / 256), 256, 0, stream>>>(
      seq_bf, w1_bf, c1b, h_bf, DH_, 9, -4, 1.0f, 1);

  // ---- conv2 (1x1) + residual + LN2 + mask -> d_out (f32)
  gemm_ln_kernel<<<dim3(MTOT / 64), 256, 0, stream>>>(
      h_bf, w2_bf, c2b, seq_f, ln2g, ln2b, mask, (float*)d_out, nullptr, DH_);
}
